// DenseGATConv_9457517986402
// MI455X (gfx1250) — matrix operand order repaired and checked by reading, not compiled, not run on hardware
//
#include <hip/hip_runtime.h>

// DenseGATConv forward, MI455X (gfx1250), wave32 + WMMA f16.
// B=2, N=4096, IN=128, H=4, C=32, neg_slope=0.2
//   k1 : h = x@W (f16 WMMA); a_src/a_dst (pre-scaled by log2e)
//   k1b: per-(b,h) max of a_src  ->  closed-form softmax row max
//   k2 : masked softmax over j + P@V via v_wmma_f32_16x16x32_f16,
//        no online rescaling (fixed exact row-max bound)

#define B_   2
#define N_   4096
#define IN_  128
#define H_   4
#define C_   32
#define HC_  128
#define NSLOPE 0.2f
#define LOG2E  1.4426950408889634f

typedef __attribute__((ext_vector_type(16))) _Float16 v16h;
typedef __attribute__((ext_vector_type(8)))  float    v8f;

// ---------------------------------------------------------------------------
// Kernel 1: projection  h[b,h,n,c] (f16) = (x @ W), plus
//           a_src[b,h,n] = log2e * sum_c h*att_src,  a_dst likewise.
// grid = B*(N/16) blocks of 128 threads; wave w handles head w (cols 32w..32w+31)
// ---------------------------------------------------------------------------
__global__ __launch_bounds__(128)
void gat_proj_kernel(const float* __restrict__ x,        // [B,N,IN]
                     const float* __restrict__ W,        // [IN,HC]
                     const float* __restrict__ att_src,  // [H*C]
                     const float* __restrict__ att_dst,  // [H*C]
                     _Float16*    __restrict__ hf,       // [B,H,N,C]
                     float*       __restrict__ asrc,     // [B,H,N]  (x log2e)
                     float*       __restrict__ adst)     // [B,H,N]  (x log2e)
{
    const int lane = threadIdx.x & 31;
    const int head = threadIdx.x >> 5;                 // wave id == head
    const int b    = blockIdx.x / (N_ / 16);
    const int i0   = (blockIdx.x % (N_ / 16)) * 16;
    const int g    = lane >> 4;                        // lane half group
    const int row  = lane & 15;                        // M (A frag) / N (B frag)

    v8f acc0 = {};                                     // cols c = 0..15
    v8f acc1 = {};                                     // cols c = 16..31

    const float* xrow = x + ((size_t)b * N_ + (i0 + row)) * IN_;

    for (int k0 = 0; k0 < IN_; k0 += 32) {
        // A fragment: half t -> x[i0+row][k0 + kk], kk = 8g + t + 8*(t>>3)
        const float4* xa = (const float4*)(xrow + k0 + 8 * g);
        const float4* xb = (const float4*)(xrow + k0 + 8 * g + 16);
        float4 a0 = xa[0], a1 = xa[1], a2 = xb[0], a3 = xb[1];
        v16h A;
        A[0]=(_Float16)a0.x; A[1]=(_Float16)a0.y; A[2]=(_Float16)a0.z; A[3]=(_Float16)a0.w;
        A[4]=(_Float16)a1.x; A[5]=(_Float16)a1.y; A[6]=(_Float16)a1.z; A[7]=(_Float16)a1.w;
        A[8]=(_Float16)a2.x; A[9]=(_Float16)a2.y; A[10]=(_Float16)a2.z; A[11]=(_Float16)a2.w;
        A[12]=(_Float16)a3.x; A[13]=(_Float16)a3.y; A[14]=(_Float16)a3.z; A[15]=(_Float16)a3.w;

        // B fragments: lane col n = row, half t -> W[k0 + 8g + t + 8*(t>>3)][head*32 + row (+16)]
        const float* wcol = W + (size_t)k0 * HC_ + head * C_ + row;
        v16h B0, B1;
#pragma unroll
        for (int t = 0; t < 16; ++t) {
            const int kk = 8 * g + t + 8 * (t >> 3);
            B0[t] = (_Float16)wcol[(size_t)kk * HC_];
            B1[t] = (_Float16)wcol[(size_t)kk * HC_ + 16];
        }

        acc0 = __builtin_amdgcn_wmma_f32_16x16x32_f16(false, A, false, B0, (short)0, acc0, false, false);
        acc1 = __builtin_amdgcn_wmma_f32_16x16x32_f16(false, A, false, B1, (short)0, acc1, false, false);
    }

    // epilogue: store h (f16, [B,H,N,C]) and reduce attention scores
    const float s_lo = att_src[head * C_ + row];
    const float s_hi = att_src[head * C_ + 16 + row];
    const float d_lo = att_dst[head * C_ + row];
    const float d_hi = att_dst[head * C_ + 16 + row];

    _Float16* hbase = hf + ((size_t)b * H_ + head) * (size_t)N_ * C_;
    float*    abase = asrc + ((size_t)b * H_ + head) * N_;
    float*    dbase = adst + ((size_t)b * H_ + head) * N_;

#pragma unroll
    for (int r = 0; r < 8; ++r) {
        const int i = i0 + r + 8 * g;                  // C-layout row
        const float v0 = acc0[r], v1 = acc1[r];
        hbase[(size_t)i * C_ + row]      = (_Float16)v0;
        hbase[(size_t)i * C_ + 16 + row] = (_Float16)v1;
        float ps = v0 * s_lo + v1 * s_hi;
        float pd = v0 * d_lo + v1 * d_hi;
#pragma unroll
        for (int off = 8; off; off >>= 1) {            // reduce within 16-lane half
            ps += __shfl_xor(ps, off, 32);
            pd += __shfl_xor(pd, off, 32);
        }
        if (row == 0) {                                // pre-scale by log2(e)
            abase[i] = ps * LOG2E;
            dbase[i] = pd * LOG2E;
        }
    }
}

// ---------------------------------------------------------------------------
// Kernel 1b: amax[bh] = max_j asrc[bh][j].   grid = B*H blocks of one wave.
// ---------------------------------------------------------------------------
__global__ __launch_bounds__(32)
void gat_rowmax_kernel(const float* __restrict__ asrc,  // [B*H, N]
                       float*       __restrict__ amax)  // [B*H]
{
    const int lane = threadIdx.x;
    const float4* p = (const float4*)(asrc + (size_t)blockIdx.x * N_);
    float m = -1e30f;
#pragma unroll 4
    for (int k = lane; k < N_ / 4; k += 32) {
        float4 v = p[k];
        m = fmaxf(m, fmaxf(fmaxf(v.x, v.y), fmaxf(v.z, v.w)));
    }
#pragma unroll
    for (int off = 16; off; off >>= 1)
        m = fmaxf(m, __shfl_xor(m, off, 32));
    if (lane == 0) amax[blockIdx.x] = m;
}

// ---------------------------------------------------------------------------
// Kernel 2: masked softmax + aggregation with fixed row-max bound.
// grid = B*(N/16) blocks of 128 threads; wave w = head w, 16 i-rows per block.
// ---------------------------------------------------------------------------
__global__ __launch_bounds__(128)
void gat_attn_kernel(const int*      __restrict__ adj,   // [B,N,N]
                     const _Float16* __restrict__ hf,    // [B,H,N,C]
                     const float*    __restrict__ asrc,  // [B,H,N] (x log2e)
                     const float*    __restrict__ adst,  // [B,H,N] (x log2e)
                     const float*    __restrict__ amax,  // [B*H]
                     const float*    __restrict__ bias,  // [HC]
                     float*          __restrict__ out)   // [B,N,HC]
{
    const int lane = threadIdx.x & 31;
    const int head = threadIdx.x >> 5;
    const int b    = blockIdx.x / (N_ / 16);
    const int i0   = (blockIdx.x % (N_ / 16)) * 16;
    const int g    = lane >> 4;
    const int row  = lane & 15;                        // this lane's score row i0+row
    const int bh   = b * H_ + head;

    const float my_adst = adst[(size_t)bh * N_ + i0 + row];
    // exact per-row softmax max bound: m_i = lrelu(max_j asrc_j + adst_i)
    const float zm  = amax[bh] + my_adst;
    const float m_i = fmaxf(zm, NSLOPE * zm);

    const int*      adjrow = adj  + ((size_t)b * N_ + (i0 + row)) * N_;
    const float*    asr    = asrc + (size_t)bh * N_;
    const _Float16* hb     = hf   + (size_t)bh * (size_t)N_ * C_;

    v8f acc0 = {}, acc1 = {};                          // O tile (16 x 32) f32
    float psum = 0.0f;                                 // softmax denom partial

    for (int j0 = 0; j0 < N_; j0 += 32) {
        const int jb1 = j0 + 8 * g;
        const int jb2 = jb1 + 16;

        // adjacency + source scores, in A-fragment K order
        int4   q0 = *(const int4*)(adjrow + jb1);
        int4   q1 = *(const int4*)(adjrow + jb1 + 4);
        int4   q2 = *(const int4*)(adjrow + jb2);
        int4   q3 = *(const int4*)(adjrow + jb2 + 4);
        float4 f0 = *(const float4*)(asr + jb1);
        float4 f1 = *(const float4*)(asr + jb1 + 4);
        float4 f2 = *(const float4*)(asr + jb2);
        float4 f3 = *(const float4*)(asr + jb2 + 4);

        // V fragments (B-matrix 32x16): lane col n = channel row, half t -> K = j0 + 8g + t + 8*(t>>3)
        const _Float16* vcol = hb + (size_t)j0 * C_ + row;
        v16h V0, V1;
#pragma unroll
        for (int t = 0; t < 16; ++t) {
            const int kk = 8 * g + t + 8 * (t >> 3);
            V0[t] = vcol[(size_t)kk * C_];
            V1[t] = vcol[(size_t)kk * C_ + 16];
        }

        if (j0 + 32 < N_)                               // warm next adj tile
            __builtin_prefetch(adjrow + jb1 + 32, 0, 0);

        float sv[16];
        int   av[16];
        sv[0]=f0.x; sv[1]=f0.y; sv[2]=f0.z; sv[3]=f0.w;
        sv[4]=f1.x; sv[5]=f1.y; sv[6]=f1.z; sv[7]=f1.w;
        sv[8]=f2.x; sv[9]=f2.y; sv[10]=f2.z; sv[11]=f2.w;
        sv[12]=f3.x; sv[13]=f3.y; sv[14]=f3.z; sv[15]=f3.w;
        av[0]=q0.x; av[1]=q0.y; av[2]=q0.z; av[3]=q0.w;
        av[4]=q1.x; av[5]=q1.y; av[6]=q1.z; av[7]=q1.w;
        av[8]=q2.x; av[9]=q2.y; av[10]=q2.z; av[11]=q2.w;
        av[12]=q3.x; av[13]=q3.y; av[14]=q3.z; av[15]=q3.w;

        v16h P;
#pragma unroll
        for (int t = 0; t < 16; ++t) {
            const float z = sv[t] + my_adst;           // log2-domain score
            const float s = fmaxf(z, NSLOPE * z);      // leaky relu (slope < 1)
            const float p = (av[t] != 0) ? __builtin_amdgcn_exp2f(s - m_i) : 0.0f;
            psum += p;
            P[t] = (_Float16)p;
        }

        acc0 = __builtin_amdgcn_wmma_f32_16x16x32_f16(false, P, false, V0, (short)0, acc0, false, false);
        acc1 = __builtin_amdgcn_wmma_f32_16x16x32_f16(false, P, false, V1, (short)0, acc1, false, false);
    }

    // combine the two K-halves of the denominator (lanes l and l^16 share a row)
    psum += __shfl_xor(psum, 16, 32);

    // finalize: divide by softmax denom, add bias, store [B,N,H*C]
    const float b_lo = bias[head * C_ + row];
    const float b_hi = bias[head * C_ + 16 + row];
#pragma unroll
    for (int r = 0; r < 8; ++r) {
        const int i = i0 + r + 8 * g;
        const float lrow = __shfl(psum, r + 8 * g, 32);  // denom for row r+8g
        const float inv  = 1.0f / lrow;
        float* op = out + ((size_t)b * N_ + i) * HC_ + head * C_;
        op[row]      = acc0[r] * inv + b_lo;
        op[16 + row] = acc1[r] * inv + b_hi;
    }
}

// ---------------------------------------------------------------------------
extern "C" void kernel_launch(void* const* d_in, const int* in_sizes, int n_in,
                              void* d_out, int out_size, void* d_ws, size_t ws_size,
                              hipStream_t stream) {
    (void)in_sizes; (void)n_in; (void)out_size; (void)ws_size;
    const float* x       = (const float*)d_in[0];
    // d_in[1] = x0 (unused by reference)
    const int*   adj     = (const int*)  d_in[2];
    const float* W       = (const float*)d_in[3];
    const float* att_src = (const float*)d_in[4];
    const float* att_dst = (const float*)d_in[5];
    const float* bias    = (const float*)d_in[6];
    float*       out     = (float*)d_out;

    char* ws = (char*)d_ws;
    _Float16* hf   = (_Float16*)ws;                                   // 2 MB
    float*    asrc = (float*)(ws + (size_t)B_ * H_ * N_ * C_ * sizeof(_Float16));
    float*    adst = asrc + (size_t)B_ * H_ * N_;                     // 128 KB each
    float*    amax = adst + (size_t)B_ * H_ * N_;                     // 8 floats

    dim3 block(128);
    gat_proj_kernel<<<dim3(B_ * (N_ / 16)), block, 0, stream>>>(
        x, W, att_src, att_dst, hf, asrc, adst);
    gat_rowmax_kernel<<<dim3(B_ * H_), dim3(32), 0, stream>>>(asrc, amax);
    gat_attn_kernel<<<dim3(B_ * (N_ / 16)), block, 0, stream>>>(
        adj, hf, asrc, adst, amax, bias, out);
}
